// CharBILSTM_37503654428750
// MI455X (gfx1250) — matrix operand order repaired and checked by reading, not compiled, not run on hardware
//
#include <hip/hip_runtime.h>

// ---------------------------------------------------------------------------
// CharBiLSTM fused kernel for MI455X (gfx1250, wave32, WMMA f16 16x16x32).
// B=64, W=128, L=32, V=128, E=128, H=256. N = B*W = 8192 words.
// One workgroup (256 threads = 8 waves) processes 16 words end-to-end.
// Weights are repacked once into WMMA B-fragment order (f16) in workspace and
// streamed from L2 every step (1.8 MB total, resident in the 192 MB L2).
// ---------------------------------------------------------------------------

#define BB   64
#define WW   128
#define LL   32
#define VV   128
#define EE   128
#define HH   256
#define G4H  1024           // 4*H
#define NWORDS (BB*WW)      // 8192
#define TILE_M 16           // words per workgroup
#define NWG   (NWORDS/TILE_M)  // 512

typedef _Float16 half8  __attribute__((ext_vector_type(8)));
typedef _Float16 half16 __attribute__((ext_vector_type(16)));
typedef float    float8 __attribute__((ext_vector_type(8)));

static __device__ __forceinline__ half16 cat88(half8 lo, half8 hi) {
  return __builtin_shufflevector(lo, hi, 0,1,2,3,4,5,6,7,8,9,10,11,12,13,14,15);
}
static __device__ __forceinline__ float8 wmma_f16(half16 a, half16 b, float8 c) {
  // 8 args: (neg_a, A, neg_b, B, c_mod, C, reuse_a, reuse_b)
  return __builtin_amdgcn_wmma_f32_16x16x32_f16(false, a, false, b, (short)0, c,
                                                false, false);
}
static __device__ __forceinline__ float sigf(float x) {
  return 1.0f / (1.0f + __expf(-x));
}
static __device__ __forceinline__ float tanh_fast(float x) {
  float e = __expf(-2.0f * x);
  return (1.0f - e) / (1.0f + e);
}

// ---------------------------------------------------------------------------
// Prep: pack a (N x K) fp32 row-major weight as WMMA B-fragments (B = W^T,
// K x N), f16.  Tile order: [nt][kt], per tile 32 lanes x 16 halves so each
// lane's fragment is one contiguous 32-byte chunk.
//   lane l holds  n = nt*16 + (l&15),  k = kt*32 + (v>>3)*16 + (l>>4)*8 + (v&7),  v=0..15
// ---------------------------------------------------------------------------
__global__ __launch_bounds__(256)
void pack_b_kernel(const float* __restrict__ src, _Float16* __restrict__ dst,
                   int N, int K) {
  int idx = blockIdx.x * 256 + threadIdx.x;      // one packed half element
  int total = N * K;
  if (idx >= total) return;
  int v    = idx & 15;
  int lane = (idx >> 4) & 31;
  int tile = idx >> 9;
  int Ktiles = K >> 5;
  int nt = tile / Ktiles;
  int kt = tile - nt * Ktiles;
  int n = nt * 16 + (lane & 15);
  int k = kt * 32 + (v >> 3) * 16 + (lane >> 4) * 8 + (v & 7);
  dst[idx] = (_Float16)src[n * K + k];
}

// ---------------------------------------------------------------------------
// Prep: embedding fp32->f16 and fused bias sums (bih + bhh) per direction.
// ---------------------------------------------------------------------------
__global__ __launch_bounds__(256)
void prep_misc_kernel(const float* __restrict__ emb, _Float16* __restrict__ emb_h,
                      const float* __restrict__ bih_f, const float* __restrict__ bhh_f,
                      float* __restrict__ bias_f,
                      const float* __restrict__ bih_b, const float* __restrict__ bhh_b,
                      float* __restrict__ bias_b) {
  int i = blockIdx.x * 256 + threadIdx.x;
  if (i < VV * EE) emb_h[i] = (_Float16)emb[i];
  if (i < G4H) {
    bias_f[i] = bih_f[i] + bhh_f[i];
    bias_b[i] = bih_b[i] + bhh_b[i];
  }
}

// ---------------------------------------------------------------------------
// Main fused kernel: embedding + input GEMM + bi-LSTM recurrence + projection.
// ---------------------------------------------------------------------------
__global__ __launch_bounds__(256)
void bilstm_main_kernel(const int*      __restrict__ char_ids,
                        const int*      __restrict__ word_lens,
                        const _Float16* __restrict__ emb_h,
                        const _Float16* __restrict__ wih_f_p,
                        const _Float16* __restrict__ wih_b_p,
                        const _Float16* __restrict__ whh_f_p,
                        const _Float16* __restrict__ whh_b_p,
                        const _Float16* __restrict__ pw_p,
                        const float*    __restrict__ bias_f,
                        const float*    __restrict__ bias_b,
                        const float*    __restrict__ pb,
                        float*          __restrict__ out) {
  __shared__ _Float16 emb_s[VV * EE];        // 32 KB f16 embedding table
  __shared__ int      ids_s[TILE_M * LL];    // 2 KB char ids for 16 words
  __shared__ int      lens_s[TILE_M];
  __shared__ _Float16 h_lin[TILE_M * HH];    // 8 KB current h (f16, row-major)
  __shared__ _Float16 cat_s[TILE_M * 2*HH];  // 16 KB [h_bwd | h_fwd]

  const int tid  = threadIdx.x;
  const int lane = tid & 31;
  const int wave = tid >> 5;                 // 0..7
  const int col  = lane & 15;                // N within a 16-wide tile
  const int hi   = lane >> 4;                // 0/1: row/K-half selector
  const int wg   = blockIdx.x;               // 0..511
  const int word0 = wg * TILE_M;

  // ---- stage embedding table + ids + lens into LDS --------------------------
  {
    const uint4* src = (const uint4*)emb_h;
    uint4*       dst = (uint4*)emb_s;
    #pragma unroll
    for (int i = 0; i < (VV * EE * 2 / 16) / 256; ++i)   // 2048 chunks / 256
      dst[tid + i * 256] = src[tid + i * 256];
  }
  for (int i = tid; i < TILE_M * LL; i += 256)
    ids_s[i] = char_ids[word0 * LL + i];
  if (tid < TILE_M) lens_s[tid] = word_lens[word0 + tid];
  __syncthreads();

  // per-lane word lengths for the 8 C-layout rows (M = hi*8 + r)
  int lenv[8];
  #pragma unroll
  for (int r = 0; r < 8; ++r) lenv[r] = lens_s[hi * 8 + r];

  #pragma unroll 1
  for (int dir = 0; dir < 2; ++dir) {
    const _Float16* wih  = dir ? wih_b_p : wih_f_p;
    const _Float16* whh  = dir ? whh_b_p : whh_f_p;
    const float*    bias = dir ? bias_b  : bias_f;

    // per-lane fused bias for the 8 accumulator tiles (gate g, subtile s)
    float bcol[8];
    #pragma unroll
    for (int g = 0; g < 4; ++g)
      #pragma unroll
      for (int s = 0; s < 2; ++s) {
        int nt = g * 16 + wave * 2 + s;
        bcol[g * 2 + s] = bias[nt * 16 + col];
      }

    float8 cst[2] = {};   // cell state, units wave*32 + s*16 + col
    float8 hst[2] = {};   // hidden state (fp32 master copy, registers)

    // zero h_lin for t=0 / t=L-1
    {
      int* hz = (int*)h_lin;
      #pragma unroll
      for (int i = 0; i < (TILE_M * HH / 2) / 256; ++i)
        hz[tid + i * 256] = 0;
    }
    __syncthreads();

    #pragma unroll 1
    for (int s = 0; s < LL; ++s) {
      // Compiler barrier: the packed-weight fragment addresses are invariant
      // across this loop; without this, LICM hoists ~96 B-fragments per wave
      // out of the loop and spills them to scratch.  With it, B streams from
      // the (L2-resident) packed-weight buffers every iteration.
      asm volatile("" ::: "memory");

      const int t = dir ? (LL - 1 - s) : s;
      float8 acc[8] = {};

      // ---- input GEMM: x_t @ Wih^T, embedding gather fused into A-frag ----
      #pragma unroll
      for (int kt = 0; kt < EE / 32; ++kt) {
        const int cid = ids_s[col * LL + t];            // char id of word `col`
        const int kb  = kt * 32 + hi * 8;
        half8 lo = *(const half8*)(emb_s + cid * EE + kb);
        half8 hv = *(const half8*)(emb_s + cid * EE + kb + 16);
        half16 a = cat88(lo, hv);
        #pragma unroll
        for (int g = 0; g < 4; ++g)
          #pragma unroll
          for (int sb = 0; sb < 2; ++sb) {
            int nt = g * 16 + wave * 2 + sb;
            half16 b = *(const half16*)(wih +
                ((size_t)(nt * (EE / 32) + kt) * 32 + lane) * 16);
            acc[g * 2 + sb] = wmma_f16(a, b, acc[g * 2 + sb]);
          }
      }

      // ---- recurrent GEMM: h @ Whh^T (A from LDS h_lin, B from L2) --------
      #pragma unroll
      for (int kt = 0; kt < HH / 32; ++kt) {
        const int kb = kt * 32 + hi * 8;
        half8 lo = *(const half8*)(h_lin + col * HH + kb);
        half8 hv = *(const half8*)(h_lin + col * HH + kb + 16);
        half16 a = cat88(lo, hv);
        #pragma unroll
        for (int g = 0; g < 4; ++g)
          #pragma unroll
          for (int sb = 0; sb < 2; ++sb) {
            int nt = g * 16 + wave * 2 + sb;
            half16 b = *(const half16*)(whh +
                ((size_t)(nt * (HH / 32) + kt) * 32 + lane) * 16);
            acc[g * 2 + sb] = wmma_f16(a, b, acc[g * 2 + sb]);
          }
      }

      __syncthreads();   // all waves done reading h_lin

      // ---- LSTM elementwise: pure-register, gate-aligned layout -----------
      #pragma unroll
      for (int sb = 0; sb < 2; ++sb) {
        #pragma unroll
        for (int r = 0; r < 8; ++r) {
          float ig = sigf(acc[0 + sb][r] + bcol[0 + sb]);
          float fg = sigf(acc[2 + sb][r] + bcol[2 + sb]);
          float gg = tanh_fast(acc[4 + sb][r] + bcol[4 + sb]);
          float og = sigf(acc[6 + sb][r] + bcol[6 + sb]);
          float cn = fg * cst[sb][r] + ig * gg;
          float hn = og * tanh_fast(cn);
          bool  m  = t < lenv[r];
          cst[sb][r] = m ? cn : cst[sb][r];
          hst[sb][r] = m ? hn : hst[sb][r];
        }
        // wave w owns exactly k-tile w of next step's A matrix
        const int unit = wave * 32 + sb * 16 + col;
        #pragma unroll
        for (int r = 0; r < 8; ++r) {
          int word = hi * 8 + r;
          h_lin[word * HH + unit] = (_Float16)hst[sb][r];
        }
      }
      __syncthreads();   // h_lin ready for next step
    }

    // final h of this direction -> concat buffer ([h_bwd | h_fwd])
    const int cbase = dir ? 0 : HH;
    #pragma unroll
    for (int sb = 0; sb < 2; ++sb) {
      const int unit = wave * 32 + sb * 16 + col;
      #pragma unroll
      for (int r = 0; r < 8; ++r) {
        int word = hi * 8 + r;
        cat_s[word * (2 * HH) + cbase + unit] = (_Float16)hst[sb][r];
      }
    }
    __syncthreads();
  }

  // ---- projection: concat[16x512] @ pW^T[512x256] + pb ----------------------
  float8 pacc[2] = {};
  #pragma unroll
  for (int kt = 0; kt < (2 * HH) / 32; ++kt) {
    const int kb = kt * 32 + hi * 8;
    half8 lo = *(const half8*)(cat_s + col * (2 * HH) + kb);
    half8 hv = *(const half8*)(cat_s + col * (2 * HH) + kb + 16);
    half16 a = cat88(lo, hv);
    #pragma unroll
    for (int sb = 0; sb < 2; ++sb) {
      int nt = wave * 2 + sb;
      half16 b = *(const half16*)(pw_p +
          ((size_t)(nt * ((2 * HH) / 32) + kt) * 32 + lane) * 16);
      pacc[sb] = wmma_f16(a, b, pacc[sb]);
    }
  }
  #pragma unroll
  for (int sb = 0; sb < 2; ++sb) {
    const int n = wave * 32 + sb * 16 + col;
    const float pbv = pb[n];
    #pragma unroll
    for (int r = 0; r < 8; ++r) {
      int word = word0 + hi * 8 + r;
      out[(size_t)word * HH + n] = pacc[sb][r] + pbv;
    }
  }

  // second tuple output: lens_out = full((B,), W) int32, appended flat
  if (wg == 0 && tid < BB)
    ((int*)out)[(size_t)NWORDS * HH + tid] = WW;
}

// ---------------------------------------------------------------------------
// Host launcher
// ---------------------------------------------------------------------------
extern "C" void kernel_launch(void* const* d_in, const int* in_sizes, int n_in,
                              void* d_out, int out_size, void* d_ws, size_t ws_size,
                              hipStream_t stream) {
  const int*   char_ids  = (const int*)  d_in[0];
  const int*   word_lens = (const int*)  d_in[1];
  const float* emb_table = (const float*)d_in[2];
  const float* Wih_f     = (const float*)d_in[3];
  const float* Whh_f     = (const float*)d_in[4];
  const float* bih_f     = (const float*)d_in[5];
  const float* bhh_f     = (const float*)d_in[6];
  const float* Wih_b     = (const float*)d_in[7];
  const float* Whh_b     = (const float*)d_in[8];
  const float* bih_b     = (const float*)d_in[9];
  const float* bhh_b     = (const float*)d_in[10];
  const float* pW        = (const float*)d_in[11];
  const float* pb        = (const float*)d_in[12];
  float*       out       = (float*)d_out;

  // workspace layout (bytes)
  char* ws = (char*)d_ws;
  _Float16* emb_h   = (_Float16*)(ws + 0);                 //  32 KB
  _Float16* wih_f_p = (_Float16*)(ws + 32768);             // 256 KB
  _Float16* wih_b_p = (_Float16*)(ws + 32768 + 262144);    // 256 KB
  _Float16* whh_f_p = (_Float16*)(ws + 557056);            // 512 KB
  _Float16* whh_b_p = (_Float16*)(ws + 1081344);           // 512 KB
  _Float16* pw_p    = (_Float16*)(ws + 1605632);           // 256 KB
  float*    bias_f  = (float*)   (ws + 1867776);           //   4 KB
  float*    bias_b  = (float*)   (ws + 1871872);           //   4 KB

  prep_misc_kernel<<<64, 256, 0, stream>>>(emb_table, emb_h,
                                           bih_f, bhh_f, bias_f,
                                           bih_b, bhh_b, bias_b);
  pack_b_kernel<<<(G4H * EE) / 256, 256, 0, stream>>>(Wih_f, wih_f_p, G4H, EE);
  pack_b_kernel<<<(G4H * EE) / 256, 256, 0, stream>>>(Wih_b, wih_b_p, G4H, EE);
  pack_b_kernel<<<(G4H * HH) / 256, 256, 0, stream>>>(Whh_f, whh_f_p, G4H, HH);
  pack_b_kernel<<<(G4H * HH) / 256, 256, 0, stream>>>(Whh_b, whh_b_p, G4H, HH);
  pack_b_kernel<<<(HH * 2 * HH) / 256, 256, 0, stream>>>(pW, pw_p, HH, 2 * HH);

  bilstm_main_kernel<<<NWG, 256, 0, stream>>>(char_ids, word_lens, emb_h,
                                              wih_f_p, wih_b_p, whh_f_p, whh_b_p,
                                              pw_p, bias_f, bias_b, pb, out);
}
